// TimeSformer_BVP_61151744360977
// MI455X (gfx1250) — matrix operand order repaired and checked by reading, not compiled, not run on hardware
//
#include <hip/hip_runtime.h>
#include <hip/hip_bf16.h>
#include <math.h>

// ---------------------------------------------------------------------------
// TimeSformer + UNet + HR head forward for MI455X (gfx1250, wave32, WMMA).
// All dense GEMMs use V_WMMA_F32_16X16X32_F16 (f16 operands, f32 accum),
// register-blocked 2x2 per wave (4 wmma per K-step). B operands (weights)
// are staged pre-transposed so both A and B fragments load as ds_load_b128.
// ---------------------------------------------------------------------------

typedef __attribute__((ext_vector_type(16))) _Float16 v16h;
typedef __attribute__((ext_vector_type(8)))  float    v8f;

#define BQ   4
#define FQ   128
#define NPAT 16          // (64/16)^2 patches per frame
#define NTOK 2049        // 1 + F*NPAT
#define DIMQ 128
#define MTOK (BQ*NTOK)   // 8196 rows
#define MTOKP 8256       // padded to 64-row block tile (129 blocks * 64)
#define MPATCH (BQ*FQ*NPAT) // 8192 rows (already multiple of 64)

// ------------------------------ WMMA GEMM ---------------------------------
// C[M,N] = A[M,K](f16) * Bt[N,K](f16, pre-transposed) (+bias)(+res)(+relu).
// Requirements (all call sites satisfy): K % 32 == 0, N % 128 == 0,
// A allocated with rows padded up to gridDim.y*64.
__global__ __launch_bounds__(256) void k_gemm_wmma(
    const _Float16* __restrict__ A, const _Float16* __restrict__ Bt,
    float* __restrict__ C, const float* __restrict__ bias,
    const float* __restrict__ residual, int M, int N, int K, int act)
{
  __shared__ _Float16 As[64][32];    // M x K tile
  __shared__ _Float16 Bs[128][40];   // N x K tile (+8 halves row pad)
  int tid  = threadIdx.x;
  int lane = tid & 31;
  int wave = tid >> 5;
  int wr = wave >> 2;                // 0..1
  int wc = wave & 3;                 // 0..3
  int m0 = blockIdx.y * 64;
  int n0 = blockIdx.x * 128;
  int half = lane >> 4;
  int l15  = lane & 15;
  // staging maps (branchless, 16B vector ops)
  int ar = tid >> 2, ac = (tid & 3) * 8;    // A: 64 rows x 32 halves
  int bn = tid >> 1, bk = (tid & 1) * 16;   // Bt: 128 rows x 32 halves
  v8f acc00 = {}, acc01 = {}, acc10 = {}, acc11 = {};
  for (int kk = 0; kk < K; kk += 32) {
    *(uint4*)&As[ar][ac] =
        *(const uint4*)&A[(size_t)(m0 + ar) * K + kk + ac];
    const _Float16* bsrc = &Bt[(size_t)(n0 + bn) * K + kk + bk];
    *(uint4*)&Bs[bn][bk]     = *(const uint4*)bsrc;
    *(uint4*)&Bs[bn][bk + 8] = *(const uint4*)(bsrc + 8);
    if (kk + 32 < K) {
      __builtin_prefetch(&A[(size_t)(m0 + ar) * K + kk + 32 + ac], 0, 3);
      __builtin_prefetch(&Bt[(size_t)(n0 + bn) * K + kk + 32 + bk], 0, 3);
    }
    __syncthreads();
    v16h a0, a1, b0, b1;
#pragma unroll
    for (int e = 0; e < 8; ++e) {
      a0[e]     = As[wr * 32 + l15][half * 16 + e];
      a0[e + 8] = As[wr * 32 + l15][half * 16 + 8 + e];
      a1[e]     = As[wr * 32 + 16 + l15][half * 16 + e];
      a1[e + 8] = As[wr * 32 + 16 + l15][half * 16 + 8 + e];
    }
#pragma unroll
    for (int e = 0; e < 16; ++e) {
      b0[e] = Bs[wc * 32 + l15][half * 16 + e];
      b1[e] = Bs[wc * 32 + 16 + l15][half * 16 + e];
    }
    acc00 = __builtin_amdgcn_wmma_f32_16x16x32_f16(false, a0, false, b0,
                                                   (short)0, acc00, false, false);
    acc01 = __builtin_amdgcn_wmma_f32_16x16x32_f16(false, a0, false, b1,
                                                   (short)0, acc01, false, false);
    acc10 = __builtin_amdgcn_wmma_f32_16x16x32_f16(false, a1, false, b0,
                                                   (short)0, acc10, false, false);
    acc11 = __builtin_amdgcn_wmma_f32_16x16x32_f16(false, a1, false, b1,
                                                   (short)0, acc11, false, false);
    __syncthreads();
  }
  // epilogue: 4 16x16 tiles; D layout lane->n=l15, vgpr r->m=r+8*half
  {
    int mb = m0 + wr * 32;
    int nb = n0 + wc * 32;
#pragma unroll
    for (int tj = 0; tj < 2; ++tj) {
      int n = nb + tj * 16 + l15;
      float bval = bias ? bias[n] : 0.f;
#pragma unroll
      for (int ti = 0; ti < 2; ++ti) {
        v8f acc = (ti == 0) ? (tj == 0 ? acc00 : acc01)
                            : (tj == 0 ? acc10 : acc11);
#pragma unroll
        for (int r = 0; r < 8; ++r) {
          int m = mb + ti * 16 + r + (half << 3);
          if (m < M) {
            float v = acc[r] + bval;
            if (residual) v += residual[(size_t)m * N + n];
            if (act == 1) v = fmaxf(v, 0.f);
            C[(size_t)m * N + n] = v;
          }
        }
      }
    }
  }
}

// --------------------------- elementwise utils -----------------------------
__global__ void k_f32_to_f16(const float* __restrict__ s, _Float16* __restrict__ d,
                             long long n) {
  long long i = (long long)blockIdx.x * 256 + threadIdx.x;
  if (i < n) d[i] = (_Float16)s[i];
}

// weight f32 [K,N] -> f16 transposed [N,K] (coalesced writes along K)
__global__ void k_f32_to_f16t(const float* __restrict__ s, _Float16* __restrict__ d,
                              int K, int N) {
  long long i = (long long)blockIdx.x * 256 + threadIdx.x;
  long long total = (long long)K * N;
  if (i >= total) return;
  int k = (int)(i % K);
  int n = (int)(i / K);
  d[(size_t)n * K + k] = (_Float16)s[(size_t)k * N + n];
}

__global__ void k_zero_h(_Float16* __restrict__ p, long long n) {
  long long i = (long long)blockIdx.x * 256 + threadIdx.x;
  if (i < n) p[i] = (_Float16)0.f;
}

__global__ void k_patchify(const float* __restrict__ video, _Float16* __restrict__ out) {
  long long i = (long long)blockIdx.x * 256 + threadIdx.x;
  long long total = (long long)MPATCH * 768;
  if (i >= total) return;
  int col = (int)(i % 768);
  long long row = i / 768;
  int b  = (int)(row / (FQ * NPAT));
  int r2 = (int)(row % (FQ * NPAT));
  int f  = r2 / NPAT;
  int pi = r2 % NPAT;
  int hy = pi >> 2, hx = pi & 3;
  int c  = col % 3;
  int pq = col / 3;
  int py = pq >> 4, px = pq & 15;
  long long src = ((((long long)b * FQ + f) * 3 + c) * 64 + (hy * 16 + py)) * 64
                  + (hx * 16 + px);
  out[i] = (_Float16)video[src];
}

__global__ void k_build_x(const float* __restrict__ tokens, const float* __restrict__ cls,
                          const float* __restrict__ pos, float* __restrict__ x) {
  long long i = (long long)blockIdx.x * 256 + threadIdx.x;
  long long total = (long long)MTOK * DIMQ;
  if (i >= total) return;
  int d  = (int)(i % DIMQ);
  int tk = (int)((i / DIMQ) % NTOK);
  int b  = (int)(i / ((long long)DIMQ * NTOK));
  float v = (tk == 0) ? cls[d]
                      : tokens[((long long)b * (FQ * NPAT) + (tk - 1)) * DIMQ + d];
  x[i] = v + pos[(long long)tk * DIMQ + d];
}

__global__ __launch_bounds__(128) void k_layernorm(
    const float* __restrict__ x, const float* __restrict__ g,
    const float* __restrict__ bt, _Float16* __restrict__ oh,
    float* __restrict__ of, int rows, long long row_stride) {
  int row = blockIdx.x;
  if (row >= rows) return;
  const float* xr = x + (long long)row * row_stride;
  int d = threadIdx.x;
  float v = xr[d];
  __shared__ float red[128];
  red[d] = v; __syncthreads();
  for (int s = 64; s > 0; s >>= 1) { if (d < s) red[d] += red[d + s]; __syncthreads(); }
  float mean = red[0] / 128.f;
  __syncthreads();
  float dv = v - mean;
  red[d] = dv * dv; __syncthreads();
  for (int s = 64; s > 0; s >>= 1) { if (d < s) red[d] += red[d + s]; __syncthreads(); }
  float var = red[0] / 128.f;
  float y = dv * rsqrtf(var + 1e-5f) * g[d] + bt[d];
  if (oh) oh[(long long)row * 128 + d] = (_Float16)y;
  if (of) of[(long long)row * 128 + d] = y;
}

__global__ void k_gate_gelu(const float* __restrict__ h, _Float16* __restrict__ out,
                            long long rows) {
  long long i = (long long)blockIdx.x * 256 + threadIdx.x;
  long long total = rows * 512;
  if (i >= total) return;
  long long row = i / 512;
  int c = (int)(i % 512);
  float a = h[row * 1024 + c];
  float gg = h[row * 1024 + 512 + c];
  float ge = 0.5f * gg * (1.f + erff(gg * 0.70710678118f));
  out[i] = (_Float16)(a * ge);
}

// ------------------------------ attention ---------------------------------
// qkv: [B, NTOK, 1536] (q:0..511, k:512..1023, v:1024..1535), per head 64 dims
__global__ __launch_bounds__(128) void k_attn_cls(const float* __restrict__ qkv,
                                                  float* __restrict__ out) {
  int bh = blockIdx.x;
  int b = bh >> 3, h = bh & 7;
  long long base = (long long)b * NTOK * 1536;
  __shared__ float qs[64];
  __shared__ float sc[NTOK];
  __shared__ float red[128];
  if (threadIdx.x < 64) qs[threadIdx.x] = qkv[base + h * 64 + threadIdx.x] * 0.125f;
  __syncthreads();
  float lmax = -3.4e38f;
  for (int j = threadIdx.x; j < NTOK; j += 128) {
    long long r = base + (long long)j * 1536 + 512 + h * 64;
    float s = 0.f;
    for (int d = 0; d < 64; ++d) s += qs[d] * qkv[r + d];
    sc[j] = s;
    lmax = fmaxf(lmax, s);
  }
  red[threadIdx.x] = lmax; __syncthreads();
  for (int s = 64; s > 0; s >>= 1) {
    if (threadIdx.x < s) red[threadIdx.x] = fmaxf(red[threadIdx.x], red[threadIdx.x + s]);
    __syncthreads();
  }
  float m = red[0]; __syncthreads();
  float ls = 0.f;
  for (int j = threadIdx.x; j < NTOK; j += 128) {
    float e = expf(sc[j] - m); sc[j] = e; ls += e;
  }
  red[threadIdx.x] = ls; __syncthreads();
  for (int s = 64; s > 0; s >>= 1) {
    if (threadIdx.x < s) red[threadIdx.x] += red[threadIdx.x + s];
    __syncthreads();
  }
  float l = red[0]; __syncthreads();
  if (threadIdx.x < 64) {
    int d = threadIdx.x;
    float a = 0.f;
    for (int j = 0; j < NTOK; ++j)
      a += sc[j] * qkv[base + (long long)j * 1536 + 1024 + h * 64 + d];
    out[((long long)b * NTOK) * 512 + h * 64 + d] = a / l;
  }
}

__global__ __launch_bounds__(128) void k_attn_time(const float* __restrict__ qkv,
                                                   float* __restrict__ out) {
  int blk = blockIdx.x;                 // 0..511 = (bh, s)
  int bh = blk >> 4;
  int s  = blk & 15;
  int b = bh >> 3, h = bh & 7;
  long long base = (long long)b * NTOK * 1536;
  int t = threadIdx.x;                  // query time 0..127
  float q[64];
  {
    long long r = base + (long long)(1 + t * NPAT + s) * 1536 + h * 64;
#pragma unroll
    for (int d = 0; d < 64; ++d) q[d] = qkv[r + d] * 0.125f;
  }
  float m = -3.4e38f, l = 0.f, acc[64];
#pragma unroll
  for (int d = 0; d < 64; ++d) acc[d] = 0.f;
  __shared__ float Ks[64][64];
  __shared__ float Vs[64][64];
  for (int c0 = 0; c0 < 129; c0 += 64) {
    int cnt = 129 - c0; if (cnt > 64) cnt = 64;
    for (int i = threadIdx.x; i < cnt * 64; i += 128) {
      int j = i >> 6, d = i & 63;
      int jj = c0 + j;
      int tok = (jj == 0) ? 0 : (1 + (jj - 1) * NPAT + s);
      long long r = base + (long long)tok * 1536;
      Ks[j][d] = qkv[r + 512 + h * 64 + d];
      Vs[j][d] = qkv[r + 1024 + h * 64 + d];
    }
    __syncthreads();
    for (int j = 0; j < cnt; ++j) {
      float sc = 0.f;
#pragma unroll
      for (int d = 0; d < 64; ++d) sc += q[d] * Ks[j][d];
      float mn = fmaxf(m, sc);
      float corr = expf(m - mn);
      float e = expf(sc - mn);
      l = l * corr + e;
#pragma unroll
      for (int d = 0; d < 64; ++d) acc[d] = acc[d] * corr + e * Vs[j][d];
      m = mn;
    }
    __syncthreads();
  }
  int tok = 1 + t * NPAT + s;
  long long o = ((long long)b * NTOK + tok) * 512 + h * 64;
  float inv = 1.f / l;
#pragma unroll
  for (int d = 0; d < 64; ++d) out[o + d] = acc[d] * inv;
}

__global__ __launch_bounds__(32) void k_attn_space(const float* __restrict__ qkv,
                                                   float* __restrict__ out) {
  int blk = blockIdx.x;                 // 0..4095 = (bh, t)
  int bh = blk >> 7;
  int t  = blk & 127;
  int b = bh >> 3, h = bh & 7;
  long long base = (long long)b * NTOK * 1536;
  __shared__ float Ks[17][64];
  __shared__ float Vs[17][64];
  for (int i = threadIdx.x; i < 17 * 64; i += 32) {
    int j = i >> 6, d = i & 63;
    int tok = (j == 0) ? 0 : (1 + t * NPAT + (j - 1));
    long long r = base + (long long)tok * 1536;
    Ks[j][d] = qkv[r + 512 + h * 64 + d];
    Vs[j][d] = qkv[r + 1024 + h * 64 + d];
  }
  __syncthreads();
  int s = threadIdx.x;
  if (s < NPAT) {
    int tok = 1 + t * NPAT + s;
    long long qr = base + (long long)tok * 1536 + h * 64;
    float q[64];
#pragma unroll
    for (int d = 0; d < 64; ++d) q[d] = qkv[qr + d] * 0.125f;
    float sc[17];
    float m = -3.4e38f;
#pragma unroll
    for (int j = 0; j < 17; ++j) {
      float v = 0.f;
#pragma unroll
      for (int d = 0; d < 64; ++d) v += q[d] * Ks[j][d];
      sc[j] = v; m = fmaxf(m, v);
    }
    float l = 0.f;
#pragma unroll
    for (int j = 0; j < 17; ++j) { sc[j] = expf(sc[j] - m); l += sc[j]; }
    long long o = ((long long)b * NTOK + tok) * 512 + h * 64;
    float inv = 1.f / l;
#pragma unroll
    for (int d = 0; d < 64; ++d) {
      float a = 0.f;
#pragma unroll
      for (int j = 0; j < 17; ++j) a += sc[j] * Vs[j][d];
      out[o + d] = a * inv;
    }
  }
}

// ------------------------------ conv / bn ---------------------------------
__global__ void k_conv3x3(const float* __restrict__ in, const float* __restrict__ w,
                          float* __restrict__ out, int Cin, int H, int W,
                          int Cout, int Ho, int Wo, int stride, long long total) {
  long long i = (long long)blockIdx.x * 256 + threadIdx.x;
  if (i >= total) return;
  int x  = (int)(i % Wo);
  int y  = (int)((i / Wo) % Ho);
  int co = (int)((i / ((long long)Wo * Ho)) % Cout);
  int b  = (int)(i / ((long long)Wo * Ho * Cout));
  float acc = 0.f;
  const float* wc = w + (long long)co * Cin * 9;
  const float* ib = in + (long long)b * Cin * H * W;
  for (int ci = 0; ci < Cin; ++ci) {
    const float* ip = ib + (long long)ci * H * W;
    const float* wp = wc + ci * 9;
    for (int ky = 0; ky < 3; ++ky) {
      int iy = y * stride + ky - 1;
      if (iy < 0 || iy >= H) continue;
      for (int kx = 0; kx < 3; ++kx) {
        int ix = x * stride + kx - 1;
        if (ix < 0 || ix >= W) continue;
        acc += ip[iy * W + ix] * wp[ky * 3 + kx];
      }
    }
  }
  out[i] = acc;
}

__global__ void k_conv1x1(const float* __restrict__ in, const float* __restrict__ w,
                          const float* __restrict__ bias, float* __restrict__ out,
                          int Cin, int H, int W, int Cout, int Ho, int Wo,
                          int stride, long long total) {
  long long i = (long long)blockIdx.x * 256 + threadIdx.x;
  if (i >= total) return;
  int x  = (int)(i % Wo);
  int y  = (int)((i / Wo) % Ho);
  int co = (int)((i / ((long long)Wo * Ho)) % Cout);
  int b  = (int)(i / ((long long)Wo * Ho * Cout));
  float acc = bias ? bias[co] : 0.f;
  const float* ib = in + (long long)b * Cin * H * W;
  int iy = y * stride, ix = x * stride;
  for (int ci = 0; ci < Cin; ++ci)
    acc += ib[(long long)ci * H * W + iy * W + ix] * w[(long long)co * Cin + ci];
  out[i] = acc;
}

__global__ __launch_bounds__(256) void k_bnstats(const float* __restrict__ x,
                                                 float* __restrict__ mean,
                                                 float* __restrict__ var,
                                                 int C, int HW, int Bn) {
  int c = blockIdx.x;
  int tid = threadIdx.x;
  float s = 0.f, s2 = 0.f;
  for (int b = 0; b < Bn; ++b) {
    const float* p = x + ((long long)b * C + c) * HW;
    for (int i = tid; i < HW; i += 256) { float v = p[i]; s += v; s2 += v * v; }
  }
  __shared__ float rs[256], rq[256];
  rs[tid] = s; rq[tid] = s2; __syncthreads();
  for (int st = 128; st > 0; st >>= 1) {
    if (tid < st) { rs[tid] += rs[tid + st]; rq[tid] += rq[tid + st]; }
    __syncthreads();
  }
  if (tid == 0) {
    float n = (float)HW * Bn;
    float m = rs[0] / n;
    mean[c] = m;
    var[c]  = rq[0] / n - m * m;
  }
}

__global__ void k_bn_act(float* __restrict__ dst, const float* __restrict__ src,
                         const float* __restrict__ mean, const float* __restrict__ var,
                         const float* __restrict__ g, const float* __restrict__ bt,
                         int C, int HW, long long total, int relu,
                         const float* __restrict__ addto) {
  long long i = (long long)blockIdx.x * 256 + threadIdx.x;
  if (i >= total) return;
  int c = (int)((i / HW) % C);
  float v = (src[i] - mean[c]) * rsqrtf(var[c] + 1e-5f) * g[c] + bt[c];
  if (addto) v += addto[i];
  if (relu) v = fmaxf(v, 0.f);
  dst[i] = v;
}

__global__ void k_maxpool2(const float* __restrict__ in, float* __restrict__ out,
                           int C, int H, int W, long long total) {
  long long i = (long long)blockIdx.x * 256 + threadIdx.x;
  if (i >= total) return;
  int Wo = W >> 1, Ho = H >> 1;
  int x = (int)(i % Wo);
  int y = (int)((i / Wo) % Ho);
  long long bc = i / ((long long)Wo * Ho);
  const float* p = in + bc * H * W;
  int iy = y * 2, ix = x * 2;
  float a = p[iy * W + ix];
  a = fmaxf(a, p[iy * W + ix + 1]);
  a = fmaxf(a, p[(iy + 1) * W + ix]);
  a = fmaxf(a, p[(iy + 1) * W + ix + 1]);
  out[i] = a;
}

__global__ void k_up2x(const float* __restrict__ in, float* __restrict__ out,
                       int C, int H, int W, long long total) {
  long long i = (long long)blockIdx.x * 256 + threadIdx.x;
  if (i >= total) return;
  int W2 = 2 * W, H2 = 2 * H;
  int ox = (int)(i % W2);
  int oy = (int)((i / W2) % H2);
  long long bc = i / ((long long)W2 * H2);
  float ys = oy * (float)(H - 1) / (float)(2 * H - 1);
  float xs = ox * (float)(W - 1) / (float)(2 * W - 1);
  int y0 = (int)floorf(ys); int y1 = min(y0 + 1, H - 1); float wy = ys - y0;
  int x0 = (int)floorf(xs); int x1 = min(x0 + 1, W - 1); float wx = xs - x0;
  const float* p = in + bc * H * W;
  float r0 = p[y0 * W + x0] * (1.f - wx) + p[y0 * W + x1] * wx;
  float r1 = p[y1 * W + x0] * (1.f - wx) + p[y1 * W + x1] * wx;
  out[i] = r0 * (1.f - wy) + r1 * wy;
}

__global__ void k_concat2(const float* __restrict__ a, int Ca,
                          const float* __restrict__ bsrc, int Cb,
                          float* __restrict__ dst, int H, int W, long long total) {
  long long i = (long long)blockIdx.x * 256 + threadIdx.x;
  if (i >= total) return;
  int Ct = Ca + Cb;
  int x = (int)(i % W);
  int y = (int)((i / W) % H);
  int c = (int)((i / ((long long)W * H)) % Ct);
  int b = (int)(i / ((long long)W * H * Ct));
  dst[i] = (c < Ca) ? a[(((long long)b * Ca + c) * H + y) * W + x]
                    : bsrc[(((long long)b * Cb + (c - Ca)) * H + y) * W + x];
}

__global__ __launch_bounds__(64) void k_avgpool(const float* __restrict__ in,
                                                float* __restrict__ out, int HW) {
  int bc = blockIdx.x;
  int tid = threadIdx.x;
  float s = 0.f;
  for (int i = tid; i < HW; i += 64) s += in[(long long)bc * HW + i];
  __shared__ float red[64];
  red[tid] = s; __syncthreads();
  for (int st = 32; st > 0; st >>= 1) {
    if (tid < st) red[tid] += red[tid + st];
    __syncthreads();
  }
  if (tid == 0) out[bc] = red[0] / (float)HW;
}

__global__ void k_small_gemm(const float* __restrict__ A, const float* __restrict__ W,
                             const float* __restrict__ bias, float* __restrict__ out,
                             int M, int K, int N, int relu) {
  int i = blockIdx.x * 256 + threadIdx.x;
  if (i >= M * N) return;
  int m = i / N, n = i % N;
  float a = bias ? bias[n] : 0.f;
  for (int k = 0; k < K; ++k) a += A[m * K + k] * W[k * N + n];
  if (relu) a = fmaxf(a, 0.f);
  out[i] = a;
}

__global__ void k_concat_vec(const float* __restrict__ a, const float* __restrict__ b,
                             float* __restrict__ z) {
  int i = blockIdx.x * 256 + threadIdx.x;
  if (i >= 4 * 256) return;
  int m = i / 256, c = i % 256;
  z[i] = (c < 128) ? a[m * 128 + c] : b[m * 128 + (c - 128)];
}

// ------------------------------ input indices ------------------------------
// jax pytree order: dicts flattened with sorted keys, recursively.
enum {
  IN_VIDEO = 0, IN_XUNET = 1, IN_CLS = 2,
  IN_DOWN1 = 3, IN_DOWN2 = 9, IN_DOWN3 = 15, IN_DOWN4 = 21,
  IN_FR_B1 = 27, IN_FR_B2 = 28, IN_FR_W1 = 29, IN_FR_W2 = 30,
  IN_HR1 = 31, IN_HR2 = 40, IN_HR3 = 46, IN_HR4 = 55,
  IN_INC = 61,
  IN_F_B1 = 67, IN_F_B2 = 68, IN_F_LN_B = 69, IN_F_LN_G = 70,
  IN_F_W1 = 71, IN_F_W2 = 72,
  IN_S_LN_B = 73, IN_S_LN_G = 74, IN_S_OB = 75, IN_S_OW = 76, IN_S_QKV = 77,
  IN_T_LN_B = 78, IN_T_LN_G = 79, IN_T_OB = 80, IN_T_OW = 81, IN_T_QKV = 82,
  IN_M_B1 = 83, IN_M_B2 = 84, IN_M_B3 = 85,
  IN_M_W1 = 86, IN_M_W2 = 87, IN_M_W3 = 88,
  IN_OUTC_B = 89, IN_OUTC_W = 90, IN_PATCH_B = 91, IN_PATCH_W = 92,
  IN_POS = 93, IN_TF_B = 94, IN_TF_G = 95,
  IN_UP1 = 96, IN_UP2 = 102, IN_UP3 = 108, IN_UP4 = 114
};
// dc sub-order  (sorted): b1,b2,g1,g2,w1,w2            -> +0..+5
// bb down order (sorted): b1,b2,db,dg,dw,g1,g2,w1,w2   -> +0..+8

static inline unsigned gblk(long long n) { return (unsigned)((n + 255) / 256); }

extern "C" void kernel_launch(void* const* d_in, const int* in_sizes, int n_in,
                              void* d_out, int out_size, void* d_ws, size_t ws_size,
                              hipStream_t stream) {
  (void)in_sizes; (void)n_in; (void)out_size; (void)ws_size;
  auto P = [&](int i) -> const float* { return (const float*)d_in[i]; };

  // ---------------- workspace bump allocator ----------------
  size_t off = 0;
  auto alloc = [&](size_t bytes) -> void* {
    off = (off + 255) & ~(size_t)255;
    void* p = (void*)((char*)d_ws + off);
    off += bytes;
    return p;
  };
  _Float16* Ah   = (_Float16*)alloc((size_t)MPATCH * 768 * 2); // patch A / LN out
  _Float16* A2h  = (_Float16*)alloc((size_t)MTOKP * 512 * 2);  // attn / gated f16
  _Float16* Wh   = (_Float16*)alloc((size_t)196608 * 2);       // weight f16 (transposed)
  float* tokens  = (float*)alloc((size_t)MPATCH * 128 * 4);
  float* xbuf    = (float*)alloc((size_t)MTOK * 128 * 4);
  float* qkvb    = (float*)alloc((size_t)MTOK * 1536 * 4);
  float* attnb   = (float*)alloc((size_t)MTOK * 512 * 4);
  float* hb      = (float*)alloc((size_t)MTOK * 1024 * 4);
  float* tfr     = (float*)alloc(4 * 128 * 4);
  float* meanb   = (float*)alloc(1024 * 4);
  float* varb    = (float*)alloc(1024 * 4);
  float* convtmp = (float*)alloc((size_t)4 * 48 * 256 * 256 * 4);
  float* midbuf  = (float*)alloc((size_t)4 * 48 * 256 * 256 * 4); // mids + upsamples
  float* x1  = (float*)alloc((size_t)4 * 48  * 256 * 256 * 4);
  float* p1  = (float*)alloc((size_t)4 * 48  * 128 * 128 * 4);
  float* x2  = (float*)alloc((size_t)4 * 96  * 128 * 128 * 4);
  float* p2  = (float*)alloc((size_t)4 * 96  * 64 * 64 * 4);
  float* x3  = (float*)alloc((size_t)4 * 192 * 64 * 64 * 4);
  float* p3  = (float*)alloc((size_t)4 * 192 * 32 * 32 * 4);
  float* x4  = (float*)alloc((size_t)4 * 384 * 32 * 32 * 4);
  float* p4  = (float*)alloc((size_t)4 * 384 * 16 * 16 * 4);
  float* x5  = (float*)alloc((size_t)4 * 384 * 16 * 16 * 4);
  float* cat1 = (float*)alloc((size_t)4 * 768 * 32 * 32 * 4);
  float* u1   = (float*)alloc((size_t)4 * 192 * 32 * 32 * 4);
  float* cat2 = (float*)alloc((size_t)4 * 384 * 64 * 64 * 4);
  float* u2   = (float*)alloc((size_t)4 * 96 * 64 * 64 * 4);
  float* cat3 = (float*)alloc((size_t)4 * 192 * 128 * 128 * 4);
  float* u3   = (float*)alloc((size_t)4 * 48 * 128 * 128 * 4);
  float* cat4 = (float*)alloc((size_t)4 * 96 * 256 * 256 * 4);
  float* u4   = (float*)alloc((size_t)4 * 48 * 256 * 256 * 4);
  float* hA = (float*)alloc((size_t)4 * 256 * 16 * 16 * 4);
  float* hB = (float*)alloc((size_t)4 * 256 * 16 * 16 * 4);
  float* hC = (float*)alloc((size_t)4 * 512 * 8 * 8 * 4);
  float* hD = (float*)alloc((size_t)4 * 512 * 8 * 8 * 4);
  float* pooled = (float*)alloc(4 * 512 * 4);
  float* sA  = (float*)alloc(4 * 256 * 4);
  float* bvp = (float*)alloc(4 * 128 * 4);
  float* zb  = (float*)alloc(4 * 256 * 4);
  float* sB  = (float*)alloc(4 * 128 * 4);
  float* sC  = (float*)alloc(4 * 64 * 4);
  float* out_f = (float*)d_out;

  auto cvt = [&](const float* s, _Float16* d, long long n) {
    k_f32_to_f16<<<gblk(n), 256, 0, stream>>>(s, d, n);
  };
  auto cvtT = [&](const float* s, _Float16* d, int K, int N) {
    k_f32_to_f16t<<<gblk((long long)K * N), 256, 0, stream>>>(s, d, K, N);
  };
  auto gemm = [&](const _Float16* A, const _Float16* Bt, float* C,
                  const float* bias, const float* res, int M, int N, int K, int act) {
    dim3 g((N + 127) / 128, (M + 63) / 64);
    k_gemm_wmma<<<g, 256, 0, stream>>>(A, Bt, C, bias, res, M, N, K, act);
  };

  // ======================= TimeSformer encoder ==========================
  {
    long long n = (long long)MPATCH * 768;
    k_patchify<<<gblk(n), 256, 0, stream>>>(P(IN_VIDEO), Ah);
    cvtT(P(IN_PATCH_W), Wh, 768, 128);
    gemm(Ah, Wh, tokens, P(IN_PATCH_B), nullptr, MPATCH, 128, 768, 0);
    long long nx = (long long)MTOK * 128;
    k_build_x<<<gblk(nx), 256, 0, stream>>>(tokens, P(IN_CLS), P(IN_POS), xbuf);
    // zero the padded A-rows (8196..8255) used by the transformer GEMMs
    k_zero_h<<<gblk((long long)(MTOKP - MTOK) * 128), 256, 0, stream>>>(
        Ah + (size_t)MTOK * 128, (long long)(MTOKP - MTOK) * 128);
    k_zero_h<<<gblk((long long)(MTOKP - MTOK) * 512), 256, 0, stream>>>(
        A2h + (size_t)MTOK * 512, (long long)(MTOKP - MTOK) * 512);
  }
  for (int i = 0; i < 12; ++i) {
    long long wq = (long long)i * 128 * 1536;
    long long wo = (long long)i * 512 * 128;
    // ---- time attention ----
    k_layernorm<<<MTOK, 128, 0, stream>>>(xbuf, P(IN_T_LN_G) + i * 128,
                                          P(IN_T_LN_B) + i * 128, Ah, nullptr,
                                          MTOK, 128);
    cvtT(P(IN_T_QKV) + wq, Wh, 128, 1536);
    gemm(Ah, Wh, qkvb, nullptr, nullptr, MTOK, 1536, 128, 0);
    k_attn_cls<<<32, 128, 0, stream>>>(qkvb, attnb);
    k_attn_time<<<512, 128, 0, stream>>>(qkvb, attnb);
    cvt(attnb, A2h, (long long)MTOK * 512);
    cvtT(P(IN_T_OW) + wo, Wh, 512, 128);
    gemm(A2h, Wh, xbuf, P(IN_T_OB) + i * 128, xbuf, MTOK, 128, 512, 0);
    // ---- space attention ----
    k_layernorm<<<MTOK, 128, 0, stream>>>(xbuf, P(IN_S_LN_G) + i * 128,
                                          P(IN_S_LN_B) + i * 128, Ah, nullptr,
                                          MTOK, 128);
    cvtT(P(IN_S_QKV) + wq, Wh, 128, 1536);
    gemm(Ah, Wh, qkvb, nullptr, nullptr, MTOK, 1536, 128, 0);
    k_attn_cls<<<32, 128, 0, stream>>>(qkvb, attnb);
    k_attn_space<<<4096, 32, 0, stream>>>(qkvb, attnb);
    cvt(attnb, A2h, (long long)MTOK * 512);
    cvtT(P(IN_S_OW) + wo, Wh, 512, 128);
    gemm(A2h, Wh, xbuf, P(IN_S_OB) + i * 128, xbuf, MTOK, 128, 512, 0);
    // ---- gated-GELU FFN ----
    k_layernorm<<<MTOK, 128, 0, stream>>>(xbuf, P(IN_F_LN_G) + i * 128,
                                          P(IN_F_LN_B) + i * 128, Ah, nullptr,
                                          MTOK, 128);
    cvtT(P(IN_F_W1) + (long long)i * 128 * 1024, Wh, 128, 1024);
    gemm(Ah, Wh, hb, P(IN_F_B1) + i * 1024, nullptr, MTOK, 1024, 128, 0);
    k_gate_gelu<<<gblk((long long)MTOK * 512), 256, 0, stream>>>(hb, A2h, MTOK);
    cvtT(P(IN_F_W2) + (long long)i * 512 * 128, Wh, 512, 128);
    gemm(A2h, Wh, xbuf, P(IN_F_B2) + i * 128, xbuf, MTOK, 128, 512, 0);
  }
  k_layernorm<<<4, 128, 0, stream>>>(xbuf, P(IN_TF_G), P(IN_TF_B), nullptr, tfr,
                                     4, (long long)NTOK * 128);

  // =============================== UNet =================================
  auto conv_bn = [&](const float* in, int Cin, int H, int W, const float* w,
                     const float* g_, const float* b_, int Cout, int stride,
                     int is1x1, int relu, const float* addto, float* dst) {
    int Ho = (H - 1) / stride + 1;
    int Wo = (W - 1) / stride + 1;
    long long tot = (long long)4 * Cout * Ho * Wo;
    if (is1x1)
      k_conv1x1<<<gblk(tot), 256, 0, stream>>>(in, w, nullptr, convtmp, Cin, H, W,
                                               Cout, Ho, Wo, stride, tot);
    else
      k_conv3x3<<<gblk(tot), 256, 0, stream>>>(in, w, convtmp, Cin, H, W,
                                               Cout, Ho, Wo, stride, tot);
    k_bnstats<<<Cout, 256, 0, stream>>>(convtmp, meanb, varb, Cout, Ho * Wo, 4);
    k_bn_act<<<gblk(tot), 256, 0, stream>>>(dst, convtmp, meanb, varb, g_, b_,
                                            Cout, Ho * Wo, tot, relu, addto);
  };
  auto dc_block = [&](const float* in, int Cin, int H, int W, int base,
                      int mid, int cout, float* dst) {
    conv_bn(in, Cin, H, W, P(base + 4), P(base + 2), P(base + 0), mid, 1, 0, 1,
            nullptr, midbuf);
    conv_bn(midbuf, mid, H, W, P(base + 5), P(base + 3), P(base + 1), cout, 1, 0, 1,
            nullptr, dst);
  };
  auto basic = [&](const float* in, int Cin, int H, int W, int base, int Cout,
                   int stride, bool res, float* dst) {
    int w1 = res ? 7 : 4, w2 = res ? 8 : 5;
    int g1 = res ? 5 : 2, g2 = res ? 6 : 3;
    int Ho = (H - 1) / stride + 1, Wo = (W - 1) / stride + 1;
    conv_bn(in, Cin, H, W, P(base + w1), P(base + g1), P(base + 0), Cout, stride,
            0, 1, nullptr, midbuf);
    conv_bn(midbuf, Cout, Ho, Wo, P(base + w2), P(base + g2), P(base + 1), Cout, 1,
            0, 0, nullptr, dst);
    if (res)
      conv_bn(in, Cin, H, W, P(base + 4), P(base + 3), P(base + 2), Cout, stride,
              1, 0, dst, dst);
  };
  auto pool = [&](const float* in, int C, int H, float* dst) {
    long long tot = (long long)4 * C * (H / 2) * (H / 2);
    k_maxpool2<<<gblk(tot), 256, 0, stream>>>(in, dst, C, H, H, tot);
  };
  auto up_cat = [&](const float* lo, int Clo, int Hlo, const float* skip, int Cs,
                    float* cat) {
    long long tu = (long long)4 * Clo * (2 * Hlo) * (2 * Hlo);
    k_up2x<<<gblk(tu), 256, 0, stream>>>(lo, midbuf, Clo, Hlo, Hlo, tu);
    long long tc = (long long)4 * (Cs + Clo) * (2 * Hlo) * (2 * Hlo);
    k_concat2<<<gblk(tc), 256, 0, stream>>>(skip, Cs, midbuf, Clo, cat,
                                            2 * Hlo, 2 * Hlo, tc);
  };

  dc_block(P(IN_XUNET), 3, 256, 256, IN_INC, 48, 48, x1);
  pool(x1, 48, 256, p1);
  dc_block(p1, 48, 128, 128, IN_DOWN1, 96, 96, x2);
  pool(x2, 96, 128, p2);
  dc_block(p2, 96, 64, 64, IN_DOWN2, 192, 192, x3);
  pool(x3, 192, 64, p3);
  dc_block(p3, 192, 32, 32, IN_DOWN3, 384, 384, x4);
  pool(x4, 384, 32, p4);
  dc_block(p4, 384, 16, 16, IN_DOWN4, 384, 384, x5);
  up_cat(x5, 384, 16, x4, 384, cat1);
  dc_block(cat1, 768, 32, 32, IN_UP1, 384, 192, u1);
  up_cat(u1, 192, 32, x3, 192, cat2);
  dc_block(cat2, 384, 64, 64, IN_UP2, 192, 96, u2);
  up_cat(u2, 96, 64, x2, 96, cat3);
  dc_block(cat3, 192, 128, 128, IN_UP3, 96, 48, u3);
  up_cat(u3, 48, 128, x1, 48, cat4);
  dc_block(cat4, 96, 256, 256, IN_UP4, 48, 48, u4);
  {
    long long tot = (long long)4 * 1 * 256 * 256;
    k_conv1x1<<<gblk(tot), 256, 0, stream>>>(u4, P(IN_OUTC_W), P(IN_OUTC_B), out_f,
                                             48, 256, 256, 1, 256, 256, 1, tot);
  }

  // ============================ HR head + MLPs ==========================
  basic(x4, 384, 32, 32, IN_HR1, 256, 2, true, hA);
  basic(hA, 256, 16, 16, IN_HR2, 256, 1, false, hB);
  basic(hB, 256, 16, 16, IN_HR3, 512, 2, true, hC);
  basic(hC, 512, 8, 8, IN_HR4, 512, 1, false, hD);
  k_avgpool<<<4 * 512, 64, 0, stream>>>(hD, pooled, 64);
  k_small_gemm<<<gblk(4 * 256), 256, 0, stream>>>(pooled, P(IN_FR_W1), P(IN_FR_B1),
                                                  sA, 4, 512, 256, 1);
  k_small_gemm<<<gblk(4 * 128), 256, 0, stream>>>(sA, P(IN_FR_W2), P(IN_FR_B2),
                                                  bvp, 4, 256, 128, 0);
  k_concat_vec<<<gblk(4 * 256), 256, 0, stream>>>(tfr, bvp, zb);
  k_small_gemm<<<gblk(4 * 128), 256, 0, stream>>>(zb, P(IN_M_W1), P(IN_M_B1),
                                                  sB, 4, 256, 128, 1);
  k_small_gemm<<<gblk(4 * 64), 256, 0, stream>>>(sB, P(IN_M_W2), P(IN_M_B2),
                                                 sC, 4, 128, 64, 1);
  k_small_gemm<<<1, 256, 0, stream>>>(sC, P(IN_M_W3), P(IN_M_B3),
                                      out_f + (long long)4 * 256 * 256, 4, 64, 1, 0);
}
